// HindiCausalLMAttention_37031208026140
// MI455X (gfx1250) — matrix operand order repaired and checked by reading, not compiled, not run on hardware
//
#include <hip/hip_runtime.h>
#include <hip/hip_bf16.h>

#define HID 2048
#define NH 16
#define HD 128
#define SEQ 4096
#define BS 2

typedef __attribute__((ext_vector_type(16))) __bf16 v16bf;
typedef __attribute__((ext_vector_type(8))) float v8f;

// ---- async global->LDS (CDNA5 GLOBAL_LOAD_ASYNC_TO_LDS_B128, ASYNCcnt) ----
#if defined(__has_builtin)
#if __has_builtin(__builtin_amdgcn_global_load_async_to_lds_b128)
#define HAVE_ASYNC_LDS 1
#endif
#endif
#ifndef HAVE_ASYNC_LDS
#define HAVE_ASYNC_LDS 0
#endif

typedef int v4i __attribute__((vector_size(16)));
typedef __attribute__((address_space(1))) v4i gv4i;
typedef __attribute__((address_space(3))) v4i lv4i;

struct B16cp { unsigned long long x, y; };

__device__ __forceinline__ void gld_async16(const unsigned short* g, unsigned short* l) {
#if HAVE_ASYNC_LDS
  __builtin_amdgcn_global_load_async_to_lds_b128((gv4i*)g, (lv4i*)l, 0, 0);
#else
  *(B16cp*)l = *(const B16cp*)g;
#endif
}

__device__ __forceinline__ void async_wait0() {
#if HAVE_ASYNC_LDS
#if __has_builtin(__builtin_amdgcn_s_wait_asynccnt)
  __builtin_amdgcn_s_wait_asynccnt(0);
#else
  asm volatile("s_wait_asynccnt 0x0" ::: "memory");
#endif
#endif
}

__device__ __forceinline__ unsigned short f2bf(float f) {
  unsigned int u = __float_as_uint(f);
  u += 0x7FFFu + ((u >> 16) & 1u);  // round-to-nearest-even
  return (unsigned short)(u >> 16);
}
__device__ __forceinline__ float bf2f(unsigned short h) {
  return __uint_as_float(((unsigned int)h) << 16);
}

__device__ __forceinline__ v8f wmma_bf16(v16bf a, v16bf b, v8f c) {
  return __builtin_amdgcn_wmma_f32_16x16x32_bf16(false, a, false, b, (short)0, c,
                                                 false, false);
}

// A fragment (16x32 bf16 MxK): lane m = lane&15, g = lane>>4; VGPR j holds the
// K pair at 2*(j&3) + 8*g + 16*(j>>2). K contiguous in memory, stride ldk.
__device__ __forceinline__ v16bf load_a_frag(const unsigned short* base, int ldk) {
  const int lane = threadIdx.x & 31;
  const int m = lane & 15, g = lane >> 4;
  union { v16bf v; unsigned int u[8]; } r;
  const unsigned short* p = base + m * ldk;
#pragma unroll
  for (int j = 0; j < 8; ++j) {
    const int kb = 2 * (j & 3) + 8 * g + 16 * (j >> 2);
    r.u[j] = *(const unsigned int*)(p + kb);
  }
  return r.v;
}

// B fragment (32x16 bf16 KxN): lane n = lane&15, g = lane>>4; VGPR j holds
// K pair 2*(j&3) + 8*g + 16*(j>>2). Stored K-innermost.
__device__ __forceinline__ v16bf load_b_frag(const unsigned short* base, int ldk) {
  const int lane = threadIdx.x & 31;
  const int n = lane & 15, g = lane >> 4;
  union { v16bf v; unsigned int u[8]; } r;
  const unsigned short* p = base + n * ldk + 8 * g;
#pragma unroll
  for (int j = 0; j < 8; ++j) r.u[j] = *(const unsigned int*)(p + 2 * (j & 3) + 16 * (j >> 2));
  return r.v;
}

// ---------------- Kernel 1: QKV projection + RoPE ----------------
// grid (32, 16, 3): x = 256-row M tile of B*S, y = head, z = q/k/v
// Q,K written [B][H][S][D]; V written transposed [B][H][D][S].
__global__ __launch_bounds__(256) void qkv_rope_kernel(
    const float* __restrict__ X,
    const float* __restrict__ Wq, const float* __restrict__ Wk,
    const float* __restrict__ Wv,
    unsigned short* __restrict__ Qr, unsigned short* __restrict__ Kr,
    unsigned short* __restrict__ Vb) {
  __shared__ unsigned short sA[256 * 32];       // [m][k]
  __shared__ unsigned short sB[128 * 32];       // [n][k] (K contiguous)
  __shared__ unsigned short sStage[256 * 128];  // result staging

  const int tid = threadIdx.x, lane = tid & 31, wid = tid >> 5;
  const int g = lane >> 4, cn = lane & 15;
  const int m0 = blockIdx.x * 256;
  const int h = blockIdx.y;
  const int n0 = h * HD;
  const int z = blockIdx.z;
  const float* W = (z == 0) ? Wq : ((z == 1) ? Wk : Wv);
  unsigned short* Out = (z == 0) ? Qr : ((z == 1) ? Kr : Vb);

  const v8f zero8 = {0.f, 0.f, 0.f, 0.f, 0.f, 0.f, 0.f, 0.f};
  v8f acc[2][8];
#pragma unroll
  for (int mi = 0; mi < 2; ++mi)
#pragma unroll
    for (int i = 0; i < 8; ++i) acc[mi][i] = zero8;

  for (int k0 = 0; k0 < HID; k0 += 32) {
    for (int i = tid; i < 8192; i += 256) {  // X tile -> bf16
      const int m = i >> 5, k = i & 31;
      sA[i] = f2bf(X[(size_t)(m0 + m) * HID + k0 + k]);
    }
    for (int i = tid; i < 4096; i += 256) {  // W tile -> bf16, [n][k]
      const int k = i >> 7, n = i & 127;
      sB[n * 32 + k] = f2bf(W[(size_t)(k0 + k) * HID + n0 + n]);
    }
    if (k0 + 32 < HID) {  // hint next K-step tiles into cache
      __builtin_prefetch(&X[(size_t)(m0 + tid) * HID + k0 + 32], 0, 1);
      __builtin_prefetch(&W[(size_t)(k0 + 32 + (tid >> 4)) * HID + n0 + (tid & 15) * 8], 0, 1);
    }
    __syncthreads();
    v16bf a0 = load_a_frag(&sA[(32 * wid) * 32], 32);
    v16bf a1 = load_a_frag(&sA[(32 * wid + 16) * 32], 32);
#pragma unroll
    for (int ni = 0; ni < 8; ++ni) {
      v16bf bfr = load_b_frag(&sB[(16 * ni) * 32], 32);
      acc[0][ni] = wmma_bf16(a0, bfr, acc[0][ni]);
      acc[1][ni] = wmma_bf16(a1, bfr, acc[1][ni]);
    }
    __syncthreads();
  }

  // stage C tiles (C layout: vgpr r -> row r + 8*g, col = lane&15)
#pragma unroll
  for (int mi = 0; mi < 2; ++mi)
#pragma unroll
    for (int ni = 0; ni < 8; ++ni) {
      float* f = (float*)&acc[mi][ni];
#pragma unroll
      for (int r = 0; r < 8; ++r)
        sStage[(32 * wid + 16 * mi + 8 * g + r) * 128 + 16 * ni + cn] = f2bf(f[r]);
    }
  __syncthreads();

  if (z == 2) {
    // V: write transposed [B][H][D][S] so attention V tiles are contiguous
    const int b = m0 >> 12;
    const int s0 = m0 & 4095;
    unsigned short* base = Out + (size_t)(b * NH + h) * HD * SEQ;
    for (int i = tid; i < 256 * 128; i += 256) {
      const int d = i >> 8, row = i & 255;  // row-contiguous global writes
      base[(size_t)d * SEQ + s0 + row] = sStage[(row << 7) + d];
    }
  } else {
    // RoPE + write [B][H][S][D]
    for (int i = tid; i < 256 * 128; i += 256) {
      const int row = i >> 7, d = i & 127;
      const int mg = m0 + row;
      const int b = mg >> 12;
      const int s = mg & 4095;
      const int d2 = d & 63;
      const float inv = __powf(10000.0f, -(float)d2 * (1.0f / 64.0f));
      float sn, cs;
      __sincosf((float)s * inv, &sn, &cs);
      const float x = bf2f(sStage[i]);
      const float xp = bf2f(sStage[(row << 7) + (d ^ 64)]);
      const float rot = (d < 64) ? -xp : xp;
      Out[((size_t)(b * NH + h) * SEQ + s) * HD + d] = f2bf(x * cs + rot * sn);
    }
  }
}

// ---------------- Kernel 2: flash attention (double-buffered async K/V) ------
// grid (32, 16, 2): x = 128-row Q tile, y = head, z = batch
__global__ __launch_bounds__(256) void attn_fa_kernel(
    const unsigned short* __restrict__ Qr, const unsigned short* __restrict__ Kr,
    const unsigned short* __restrict__ Vb, unsigned short* __restrict__ At) {
  __shared__ unsigned short sK[2][64 * 128];  // K tile [kv][d]; reused as P [m][kv]
  __shared__ unsigned short sV[2][128 * 64];  // V tile [d][kv] (global already [d][s])

  const int tid = threadIdx.x, lane = tid & 31, wid = tid >> 5;
  const int g = lane >> 4, cn = lane & 15;
  const int q0 = blockIdx.x * 128;
  const int h = blockIdx.y, b = blockIdx.z;
  const unsigned short* Qh = Qr + (size_t)(b * NH + h) * SEQ * HD;
  const unsigned short* Kh = Kr + (size_t)(b * NH + h) * SEQ * HD;
  const unsigned short* Vh = Vb + (size_t)(b * NH + h) * HD * SEQ;  // [d][s]

  auto issue_tiles = [&](int jb, int buf) {
    const int kv0 = jb * 64;
    for (int p = tid; p < 1024; p += 256) {  // K tile: 64x128 bf16, 16B packets
      const int kv = p >> 4, doff = (p & 15) * 8;
      gld_async16(Kh + (size_t)(kv0 + kv) * HD + doff, &sK[buf][kv * 128 + doff]);
    }
    for (int p = tid; p < 1024; p += 256) {  // V tile: 128x64 bf16, 16B packets
      const int d = p >> 3, kvoff = (p & 7) * 8;
      gld_async16(Vh + (size_t)d * SEQ + kv0 + kvoff, &sV[buf][d * 64 + kvoff]);
    }
  };

  // Q fragments straight from global (A layout, K-pairs contiguous in [s][d])
  v16bf qf[4];
#pragma unroll
  for (int kc = 0; kc < 4; ++kc)
    qf[kc] = load_a_frag(Qh + (size_t)(q0 + 16 * wid) * HD + 32 * kc, HD);

  const v8f zero8 = {0.f, 0.f, 0.f, 0.f, 0.f, 0.f, 0.f, 0.f};
  v8f o[8];
#pragma unroll
  for (int i = 0; i < 8; ++i) o[i] = zero8;
  float mrow[8], lrow[8];
#pragma unroll
  for (int r = 0; r < 8; ++r) { mrow[r] = -3.0e38f; lrow[r] = 0.f; }

  const float scale = 0.088388347648318447f;  // 1/sqrt(128)
  const int nkv = (q0 + 128) >> 6;            // causal: skip fully masked blocks

  issue_tiles(0, 0);
  async_wait0();
  __syncthreads();

  for (int jb = 0; jb < nkv; ++jb) {
    const int cur = jb & 1;
    const int kv0 = jb * 64;
    if (jb + 1 < nkv) issue_tiles(jb + 1, cur ^ 1);  // prefetch next block

    // scores: 16 rows (wave) x 64 kv
    v8f sc[4];
#pragma unroll
    for (int ni = 0; ni < 4; ++ni) sc[ni] = zero8;
#pragma unroll
    for (int kc = 0; kc < 4; ++kc) {
#pragma unroll
      for (int ni = 0; ni < 4; ++ni) {
        v16bf bfr = load_b_frag(&sK[cur][(16 * ni) * 128 + 32 * kc], 128);
        sc[ni] = wmma_bf16(qf[kc], bfr, sc[ni]);
      }
    }

    // scale + causal mask + row max
    float rmax[8];
#pragma unroll
    for (int r = 0; r < 8; ++r) rmax[r] = -3.0e38f;
#pragma unroll
    for (int ni = 0; ni < 4; ++ni) {
      float* f = (float*)&sc[ni];
      const int kcol = kv0 + 16 * ni + cn;
#pragma unroll
      for (int r = 0; r < 8; ++r) {
        const int qrow = q0 + 16 * wid + 8 * g + r;
        float v = f[r] * scale;
        if (kcol > qrow) v += -1.0e9f;
        f[r] = v;
        rmax[r] = fmaxf(rmax[r], v);
      }
    }
#pragma unroll
    for (int r = 0; r < 8; ++r)
      for (int off = 1; off < 16; off <<= 1)
        rmax[r] = fmaxf(rmax[r], __shfl_xor(rmax[r], off, 16));

    // online softmax update
    float corr[8];
#pragma unroll
    for (int r = 0; r < 8; ++r) {
      const float mnew = fmaxf(mrow[r], rmax[r]);
      corr[r] = __expf(mrow[r] - mnew);
      mrow[r] = mnew;
      lrow[r] *= corr[r];
    }
#pragma unroll
    for (int di = 0; di < 8; ++di) {
      float* f = (float*)&o[di];
#pragma unroll
      for (int r = 0; r < 8; ++r) f[r] *= corr[r];
    }
    float rsum[8];
#pragma unroll
    for (int r = 0; r < 8; ++r) rsum[r] = 0.f;
#pragma unroll
    for (int ni = 0; ni < 4; ++ni) {
      float* f = (float*)&sc[ni];
#pragma unroll
      for (int r = 0; r < 8; ++r) {
        const float p = __expf(f[r] - mrow[r]);
        f[r] = p;
        rsum[r] += p;
      }
    }
#pragma unroll
    for (int r = 0; r < 8; ++r) {
      for (int off = 1; off < 16; off <<= 1)
        rsum[r] += __shfl_xor(rsum[r], off, 16);
      lrow[r] += rsum[r];
    }

    __syncthreads();               // all waves done reading sK[cur]
    unsigned short* sP = sK[cur];  // reuse as P [m=128][kv=64]
#pragma unroll
    for (int ni = 0; ni < 4; ++ni) {
      float* f = (float*)&sc[ni];
#pragma unroll
      for (int r = 0; r < 8; ++r)
        sP[(16 * wid + 8 * g + r) * 64 + 16 * ni + cn] = f2bf(f[r]);
    }
    __syncthreads();

    // O += P @ V
#pragma unroll
    for (int kk = 0; kk < 2; ++kk) {
      v16bf pa = load_a_frag(&sP[(16 * wid) * 64 + 32 * kk], 64);
#pragma unroll
      for (int di = 0; di < 8; ++di) {
        v16bf vb = load_b_frag(&sV[cur][(16 * di) * 64 + 32 * kk], 64);
        o[di] = wmma_bf16(pa, vb, o[di]);
      }
    }
    async_wait0();   // next block's tiles landed
    __syncthreads();
  }

  // normalize + store to [B][S][H*D]
#pragma unroll
  for (int di = 0; di < 8; ++di) {
    float* f = (float*)&o[di];
#pragma unroll
    for (int r = 0; r < 8; ++r) {
      const int s = q0 + 16 * wid + 8 * g + r;
      const int d = 16 * di + cn;
      At[((size_t)(b * SEQ + s)) * HID + h * HD + d] = f2bf(f[r] / lrow[r]);
    }
  }
}

// ---------------- Kernel 3: output projection ----------------
// grid (32, 16): 256x128 output tiles, fp32 out
__global__ __launch_bounds__(256) void oproj_kernel(
    const unsigned short* __restrict__ A, const float* __restrict__ Wo,
    float* __restrict__ Out) {
  __shared__ unsigned short sA[256 * 32];
  __shared__ unsigned short sB[128 * 32];
  const int tid = threadIdx.x, lane = tid & 31, wid = tid >> 5;
  const int g = lane >> 4, cn = lane & 15;
  const int m0 = blockIdx.x * 256;
  const int n0 = blockIdx.y * 128;

  const v8f zero8 = {0.f, 0.f, 0.f, 0.f, 0.f, 0.f, 0.f, 0.f};
  v8f acc[2][8];
#pragma unroll
  for (int mi = 0; mi < 2; ++mi)
#pragma unroll
    for (int i = 0; i < 8; ++i) acc[mi][i] = zero8;

  for (int k0 = 0; k0 < HID; k0 += 32) {
    for (int p = tid; p < 1024; p += 256) {  // A tile bf16: async 16B packets
      const int m = p >> 2, koff = (p & 3) * 8;
      gld_async16(A + (size_t)(m0 + m) * HID + k0 + koff, &sA[m * 32 + koff]);
    }
    for (int i = tid; i < 4096; i += 256) {
      const int k = i >> 7, n = i & 127;
      sB[n * 32 + k] = f2bf(Wo[(size_t)(k0 + k) * HID + n0 + n]);
    }
    if (k0 + 32 < HID) {
      __builtin_prefetch(&A[(size_t)(m0 + tid) * HID + k0 + 32], 0, 1);
      __builtin_prefetch(&Wo[(size_t)(k0 + 32 + (tid >> 4)) * HID + n0 + (tid & 15) * 8], 0, 1);
    }
    async_wait0();
    __syncthreads();
    v16bf a0 = load_a_frag(&sA[(32 * wid) * 32], 32);
    v16bf a1 = load_a_frag(&sA[(32 * wid + 16) * 32], 32);
#pragma unroll
    for (int ni = 0; ni < 8; ++ni) {
      v16bf bfr = load_b_frag(&sB[(16 * ni) * 32], 32);
      acc[0][ni] = wmma_bf16(a0, bfr, acc[0][ni]);
      acc[1][ni] = wmma_bf16(a1, bfr, acc[1][ni]);
    }
    __syncthreads();
  }
#pragma unroll
  for (int mi = 0; mi < 2; ++mi)
#pragma unroll
    for (int ni = 0; ni < 8; ++ni) {
      float* f = (float*)&acc[mi][ni];
#pragma unroll
      for (int r = 0; r < 8; ++r)
        Out[(size_t)(m0 + 32 * wid + 16 * mi + 8 * g + r) * HID + n0 + 16 * ni + cn] = f[r];
    }
}

extern "C" void kernel_launch(void* const* d_in, const int* in_sizes, int n_in,
                              void* d_out, int out_size, void* d_ws, size_t ws_size,
                              hipStream_t stream) {
  const float* X  = (const float*)d_in[0];
  // d_in[1] attention_mask (causal mask recomputed analytically), d_in[2] position_ids
  const float* Wq = (const float*)d_in[3];
  const float* Wk = (const float*)d_in[4];
  const float* Wv = (const float*)d_in[5];
  const float* Wo = (const float*)d_in[6];
  float* out = (float*)d_out;

  const size_t per = (size_t)BS * NH * SEQ * HD;  // 16,777,216 bf16 elems each
  unsigned short* Qr = (unsigned short*)d_ws;
  unsigned short* Kr = Qr + per;
  unsigned short* Vb = Kr + per;   // V stored transposed [B][H][D][S]
  unsigned short* At = Vb + per;   // total ws use: 128 MiB

  dim3 blk(256);
  qkv_rope_kernel<<<dim3(32, 16, 3), blk, 0, stream>>>(X, Wq, Wk, Wv, Qr, Kr, Vb);
  attn_fa_kernel<<<dim3(32, 16, 2), blk, 0, stream>>>(Qr, Kr, Vb, At);
  oproj_kernel<<<dim3(32, 16), blk, 0, stream>>>(At, Wo, out);
}
